// SetAbstraction_45174466019645
// MI455X (gfx1250) — matrix operand order repaired and checked by reading, not compiled, not run on hardware
//
#include <hip/hip_runtime.h>

typedef __attribute__((ext_vector_type(16))) _Float16 v16h;
typedef __attribute__((ext_vector_type(8)))  _Float16 v8h;
typedef __attribute__((ext_vector_type(8)))  float    v8f;

#define B_       16
#define N_       4096
#define NPOINT_  1024
#define NSAMPLE_ 32
#define CIN_     64
#define R2_      0.04f      // radius^2 = 0.2^2
#define XS_      96         // LDS stride (f16) for activations, K padded 67 -> 96
#define ROWS_    64         // rows (pixels) per block = 2 groups of 32 samples

// ---------------------------------------------------------------------------
// Kernel 1: Farthest Point Sampling. One block per batch. 256 threads,
// each owns 16 points in registers. 1024 dependent argmax rounds (inherent).
// ---------------------------------------------------------------------------
__global__ void __launch_bounds__(256)
fps_kernel(const float* __restrict__ xyz,
           float* __restrict__ ws_newxyz,
           float* __restrict__ out_newxyz,
           int*   __restrict__ fps_idx)
{
    const int b   = blockIdx.x;
    const int tid = threadIdx.x;
    const float* X = xyz + (size_t)b * N_ * 3;

    float px[16], py[16], pz[16], dist[16];
#pragma unroll
    for (int i = 0; i < 16; ++i) {
        const int p = tid + i * 256;
        px[i] = X[p * 3 + 0];
        py[i] = X[p * 3 + 1];
        pz[i] = X[p * 3 + 2];
        dist[i] = 1e10f;
    }

    __shared__ float sd[256];
    __shared__ int   si[256];
    __shared__ float c3[3];

    int far = 0;
    for (int it = 0; it < NPOINT_; ++it) {
        if (tid == 0) {
            fps_idx[b * NPOINT_ + it] = far;
            const float cx = X[far * 3 + 0];
            const float cy = X[far * 3 + 1];
            const float cz = X[far * 3 + 2];
            c3[0] = cx; c3[1] = cy; c3[2] = cz;
            const size_t o = ((size_t)b * NPOINT_ + it) * 3;
            ws_newxyz[o + 0] = cx; ws_newxyz[o + 1] = cy; ws_newxyz[o + 2] = cz;
            out_newxyz[o + 0] = cx; out_newxyz[o + 1] = cy; out_newxyz[o + 2] = cz;
        }
        __syncthreads();
        const float cx = c3[0], cy = c3[1], cz = c3[2];

        float bd = -1.0f;
        int   bi = 0x7fffffff;
#pragma unroll
        for (int i = 0; i < 16; ++i) {
            const float dx = px[i] - cx, dy = py[i] - cy, dz = pz[i] - cz;
            const float d  = dx * dx + dy * dy + dz * dz;
            dist[i] = fminf(dist[i], d);
            const int p = tid + i * 256;
            if (dist[i] > bd || (dist[i] == bd && p < bi)) { bd = dist[i]; bi = p; }
        }
        sd[tid] = bd; si[tid] = bi;
        __syncthreads();
        for (int s = 128; s > 0; s >>= 1) {
            if (tid < s) {
                const float od = sd[tid + s];
                const int   oi = si[tid + s];
                if (od > sd[tid] || (od == sd[tid] && oi < si[tid])) { sd[tid] = od; si[tid] = oi; }
            }
            __syncthreads();
        }
        far = si[0];
    }
}

// ---------------------------------------------------------------------------
// Kernel 2: Ball query (first NSAMPLE neighbors within radius, pad w/ first).
// Thread per center.
// ---------------------------------------------------------------------------
__global__ void __launch_bounds__(256)
ballq_kernel(const float* __restrict__ xyz,
             const float* __restrict__ ws_newxyz,
             int* __restrict__ ballidx)
{
    const int c = blockIdx.x * blockDim.x + threadIdx.x; // 0 .. B*NPOINT-1
    if (c >= B_ * NPOINT_) return;
    const int b = c >> 10;

    const float cx = ws_newxyz[c * 3 + 0];
    const float cy = ws_newxyz[c * 3 + 1];
    const float cz = ws_newxyz[c * 3 + 2];
    const float* X = xyz + (size_t)b * N_ * 3;
    int* out = ballidx + (size_t)c * NSAMPLE_;

    int cnt = 0, first = -1;
    for (int j = 0; j < N_; ++j) {
        const float dx = X[j * 3 + 0] - cx;
        const float dy = X[j * 3 + 1] - cy;
        const float dz = X[j * 3 + 2] - cz;
        const float d  = dx * dx + dy * dy + dz * dz;
        if (d <= R2_) {
            if (first < 0) first = j;
            out[cnt] = j;
            if (++cnt == NSAMPLE_) break;
        }
    }
    if (first < 0) first = 0;
    for (int k = cnt; k < NSAMPLE_; ++k) out[k] = first;
}

// ---------------------------------------------------------------------------
// Kernel 3: gather + 3-layer MLP (WMMA f16, f32 accum, BN folded) + maxpool.
// ---------------------------------------------------------------------------
__device__ inline v16h cat8(v8h lo, v8h hi)
{
    v16h r;
#pragma unroll
    for (int i = 0; i < 8; ++i) { r[i] = lo[i]; r[8 + i] = hi[i]; }
    return r;
}

__device__ inline void stage_weights(const float* __restrict__ W,
                                     const float* __restrict__ bb,
                                     const float* __restrict__ g,
                                     const float* __restrict__ beta,
                                     const float* __restrict__ m,
                                     const float* __restrict__ v,
                                     int O, int K,
                                     _Float16* sW, float* sScale, float* sBias,
                                     int tid)
{
    for (int e = tid; e < O * XS_; e += 256) {
        const int o = e / XS_, k = e - o * XS_;
        sW[e] = (k < K) ? (_Float16)W[o * K + k] : (_Float16)0.f;
    }
    for (int o = tid; o < O; o += 256) {
        const float s = g[o] * rsqrtf(v[o] + 1e-5f);
        sScale[o] = s;
        sBias[o]  = (bb[o] - m[o]) * s + beta[o];
    }
}

// One layer: X (64 x Kpad, f16 in LDS) @ W^T (Kpad x OC) -> relu(scale*acc+bias)
// A-frag: 16x32 f16  lane<16: K {0..7,16..23}; lane>=16: K {8..15,24..31}
// B-frag: 32x16 f16  lane<16: K {0..7,16..23}; lane>=16: K {8..15,24..31} (N = lane&15)
template <int OC, int KSTEPS, bool LAST>
__device__ inline void mlp_layer(const _Float16* __restrict__ sX,
                                 const _Float16* __restrict__ sW,
                                 const float* __restrict__ sScale,
                                 const float* __restrict__ sBias,
                                 _Float16* __restrict__ sXout,
                                 int* __restrict__ sPool,
                                 int tid)
{
    const int wave = tid >> 5;
    const int lane = tid & 31;
    const int half = lane >> 4;
    const int r    = lane & 15;

    constexpr int NT  = OC / 16;        // N tiles (output-channel tiles)
    constexpr int TOT = 4 * NT;         // 4 M-tiles x NT
    constexpr int TPW = TOT / 8;        // tiles per wave (2 or 4)

    v8f accs[TPW];

#pragma unroll
    for (int i = 0; i < TPW; ++i) {
        const int t  = wave + i * 8;
        const int mt = t & 3;
        const int nt = t >> 2;
        v8f acc = {};
#pragma unroll
        for (int ks = 0; ks < KSTEPS; ++ks) {
            const int kb = ks * 32;
            const _Float16* ab = &sX[(mt * 16 + r) * XS_ + kb + half * 8];
            const v8h a0 = *(const v8h*)ab;          // K: kb + half*8 .. +7
            const v8h a1 = *(const v8h*)(ab + 16);   // K: kb + 16 + half*8 .. +7
            const _Float16* bp = &sW[(nt * 16 + r) * XS_ + kb + half * 8];
            const v8h b0 = *(const v8h*)bp;          // K: kb + half*8 .. +7
            const v8h b1 = *(const v8h*)(bp + 16);   // K: kb + 16 + half*8 .. +7
            const v16h A  = cat8(a0, a1);
            const v16h Bm = cat8(b0, b1);
            acc = __builtin_amdgcn_wmma_f32_16x16x32_f16(
                false, A, false, Bm, (short)0, acc, false, false);
        }
        if constexpr (LAST) {
            const int col = nt * 16 + r;     // output channel
            const float sc = sScale[col], bi = sBias[col];
            const int   grp = mt >> 1;       // which of the 2 sample-groups
#pragma unroll
            for (int vv = 0; vv < 8; ++vv) {
                const float val = fmaxf(acc[vv] * sc + bi, 0.f);
                // relu >= 0 => int-bit compare is order-preserving
                atomicMax(&sPool[grp * 128 + col], __float_as_int(val));
            }
        } else {
            accs[i] = acc;
        }
    }

    if constexpr (!LAST) {
        __syncthreads();                     // all reads of sX done before overwrite
#pragma unroll
        for (int i = 0; i < TPW; ++i) {
            const int t  = wave + i * 8;
            const int mt = t & 3;
            const int nt = t >> 2;
            const int col = nt * 16 + r;
            const float sc = sScale[col], bi = sBias[col];
#pragma unroll
            for (int vv = 0; vv < 8; ++vv) {
                const int row = mt * 16 + half * 8 + vv;   // C/D layout: M = half*8+vgpr
                const float val = fmaxf(accs[i][vv] * sc + bi, 0.f);
                sXout[row * XS_ + col] = (_Float16)val;
            }
        }
    }
    __syncthreads();
}

__global__ void __launch_bounds__(256)
sa_mlp_kernel(const float* __restrict__ xyz,
              const float* __restrict__ feat,
              const float* __restrict__ ws_newxyz,
              const int*   __restrict__ ballidx,
              const float* __restrict__ w0, const float* __restrict__ b0,
              const float* __restrict__ g0, const float* __restrict__ bt0,
              const float* __restrict__ m0, const float* __restrict__ v0,
              const float* __restrict__ w1, const float* __restrict__ b1,
              const float* __restrict__ g1, const float* __restrict__ bt1,
              const float* __restrict__ m1, const float* __restrict__ v1,
              const float* __restrict__ w2, const float* __restrict__ b2,
              const float* __restrict__ g2, const float* __restrict__ bt2,
              const float* __restrict__ m2, const float* __restrict__ v2,
              float* __restrict__ out_feat)
{
    __shared__ __align__(16) _Float16 sX[ROWS_ * XS_];   // 12 KB * 2
    __shared__ __align__(16) _Float16 sW[128 * XS_];     // 24 KB
    __shared__ float sScale[128];
    __shared__ float sBias[128];
    __shared__ int   sPool[2 * 128];

    const int tid = threadIdx.x;

    // ---- gather concat(dxyz, features) as f16, K padded [67,96) = 0 ----
    for (int e = tid; e < ROWS_ * XS_; e += 256) {
        const int row = e / XS_;
        const int k   = e - row * XS_;
        const int gid = blockIdx.x * 2 + (row >> 5);   // b*NPOINT + m
        const int s   = row & 31;
        const int b   = gid >> 10;
        const int j   = ballidx[(size_t)gid * NSAMPLE_ + s];
        float val = 0.f;
        if (k < 3)
            val = xyz[((size_t)b * N_ + j) * 3 + k] - ws_newxyz[(size_t)gid * 3 + k];
        else if (k < 3 + CIN_)
            val = feat[((size_t)b * N_ + j) * CIN_ + (k - 3)];
        sX[e] = (_Float16)val;
    }
    stage_weights(w0, b0, g0, bt0, m0, v0, 64, 67, sW, sScale, sBias, tid);
    __syncthreads();

    mlp_layer<64, 3, false>(sX, sW, sScale, sBias, sX, sPool, tid);   // 67(pad96)->64

    stage_weights(w1, b1, g1, bt1, m1, v1, 64, 64, sW, sScale, sBias, tid);
    __syncthreads();

    mlp_layer<64, 2, false>(sX, sW, sScale, sBias, sX, sPool, tid);   // 64->64

    stage_weights(w2, b2, g2, bt2, m2, v2, 128, 64, sW, sScale, sBias, tid);
    sPool[tid] = 0;                      // 256 threads, 256 pool cells (= 0.0f)
    __syncthreads();

    mlp_layer<128, 2, true>(sX, sW, sScale, sBias, sX, sPool, tid);   // 64->128 + pool

    // ---- write pooled features ----
    const int gid = blockIdx.x * 2 + (tid >> 7);
    out_feat[(size_t)gid * 128 + (tid & 127)] = __int_as_float(sPool[tid]);
}

// ---------------------------------------------------------------------------
extern "C" void kernel_launch(void* const* d_in, const int* in_sizes, int n_in,
                              void* d_out, int out_size, void* d_ws, size_t ws_size,
                              hipStream_t stream)
{
    const float* xyz  = (const float*)d_in[0];
    const float* feat = (const float*)d_in[1];
    const float* W[3], *Bb[3], *G[3], *Bt[3], *M[3], *V[3];
    for (int l = 0; l < 3; ++l) {
        W[l]  = (const float*)d_in[2 + 6 * l + 0];
        Bb[l] = (const float*)d_in[2 + 6 * l + 1];
        G[l]  = (const float*)d_in[2 + 6 * l + 2];
        Bt[l] = (const float*)d_in[2 + 6 * l + 3];
        M[l]  = (const float*)d_in[2 + 6 * l + 4];
        V[l]  = (const float*)d_in[2 + 6 * l + 5];
    }

    float* out_newxyz = (float*)d_out;                          // (16,1024,3)
    float* out_feat   = (float*)d_out + B_ * NPOINT_ * 3;       // (16,1024,128)

    // workspace: new_xyz copy | fps idx | ball idx  (~2.3 MB)
    float* ws_newxyz = (float*)d_ws;
    int*   ws_fps    = (int*)(ws_newxyz + B_ * NPOINT_ * 3);
    int*   ws_ball   = ws_fps + B_ * NPOINT_;

    fps_kernel<<<B_, 256, 0, stream>>>(xyz, ws_newxyz, out_newxyz, ws_fps);

    ballq_kernel<<<(B_ * NPOINT_) / 256, 256, 0, stream>>>(xyz, ws_newxyz, ws_ball);

    sa_mlp_kernel<<<(B_ * NPOINT_) / 2, 256, 0, stream>>>(
        xyz, feat, ws_newxyz, ws_ball,
        W[0], Bb[0], G[0], Bt[0], M[0], V[0],
        W[1], Bb[1], G[1], Bt[1], M[1], V[1],
        W[2], Bb[2], G[2], Bt[2], M[2], V[2],
        out_feat);
}
